// GatedSparseConvUNet_89232240542137
// MI455X (gfx1250) — matrix operand order repaired and checked by reading, not compiled, not run on hardware
//
#include <hip/hip_runtime.h>
#include <hip/hip_bf16.h>
#include <stdint.h>

// ---------------------------------------------------------------------------
// Types for CDNA5 WMMA (wave32): bf16 16x16x32 -> f32 accumulate
// ---------------------------------------------------------------------------
typedef __attribute__((ext_vector_type(16))) __bf16        bf16x16;
typedef __attribute__((ext_vector_type(8)))  float         f32x8;
typedef __attribute__((ext_vector_type(4)))  unsigned int  u32x4;

__device__ __forceinline__ unsigned short f2bf(float x) {
  unsigned int u = __float_as_uint(x);
  unsigned int r = (u + 0x7FFFu + ((u >> 16) & 1u)) >> 16;  // RNE
  return (unsigned short)r;
}
__device__ __forceinline__ float bf2f(unsigned short h) {
  return __uint_as_float(((unsigned int)h) << 16);
}
__device__ __forceinline__ float gate_f(float a, float g) {
  return (a > 0.f ? a : 0.1f * a) * (1.f / (1.f + __expf(-g)));
}

union ABits { u32x4 q[2]; bf16x16 v; };

// ---------------------------------------------------------------------------
// Repack f32 weights [KNBR, Csrc, TwoD] -> bf16 WMMA B-fragment chunks.
// Chunk = (k*nCt + ctile)*nColT + coltile ; within a chunk: lane holds
// column (lane&15), K-local ((j&8)<<1) + (lane>>4)*8 + (j&7), j contiguous -> one 32B load.
// Rows c >= Csrc are zero (channel padding, used for IN_DIM=16 -> 32).
// ---------------------------------------------------------------------------
__global__ void repack_w(const float* __restrict__ src, unsigned short* __restrict__ dst,
                         int KNBR, int Csrc, int Cpad, int TwoD) {
  long total = (long)KNBR * Cpad * TwoD;
  long tid = (long)blockIdx.x * blockDim.x + threadIdx.x;
  if (tid >= total) return;
  int  j    = (int)(tid & 15);
  int  lane = (int)((tid >> 4) & 31);
  long chnk = tid >> 9;
  int nColT = TwoD >> 4;
  int nCt   = Cpad >> 5;
  int colt  = (int)(chnk % nColT);
  long rem  = chnk / nColT;
  int ctile = (int)(rem % nCt);
  int k     = (int)(rem / nCt);
  int nn    = lane & 15;
  int half  = lane >> 4;
  int c     = ctile * 32 + ((j & 8) << 1) + half * 8 + (j & 7);
  int col   = colt * 16 + nn;
  float v = (c < Csrc) ? src[((long)k * Csrc + c) * TwoD + col] : 0.f;
  dst[tid] = f2bf(v);
}

// f32 [N,Csrc] -> bf16 [N,Cdst], zero-padded channels
__global__ void cvt_pad(const float* __restrict__ src, unsigned short* __restrict__ dst,
                        long N, int Csrc, int Cdst) {
  long total = N * Cdst;
  long tid = (long)blockIdx.x * blockDim.x + threadIdx.x;
  if (tid >= total) return;
  long row = tid / Cdst;
  int  c   = (int)(tid % Cdst);
  dst[tid] = f2bf(c < Csrc ? src[row * Csrc + c] : 0.f);
}

// ---------------------------------------------------------------------------
// Gathered gated conv: out[n,d] = gate( sum_k sum_c F[rule(n,k),c] W[k,c,d] )
// MODE 0: rule(n,k) = nbr[n*KNBR+k]        (27-nbr submanifold / 8-child down)
// MODE 1: rule(n,k) = (kposes[n]==k)?parents[n]:-1   (octant deconv up)
// One wave per (16-row tile, TWO column pairs): A fragment (the gathered,
// expensive operand) is reused across 4 WMMAs per 32-channel chunk.
// ---------------------------------------------------------------------------
template <int MODE>
__global__ __launch_bounds__(256) void gconv_wmma(
    const unsigned short* __restrict__ f1, int C1,
    const unsigned short* __restrict__ f2, int C2,
    const int* __restrict__ nbr,
    const int* __restrict__ parents,
    const int* __restrict__ kposes,
    const unsigned short* __restrict__ Bp,
    const unsigned short* __restrict__ resid,
    unsigned short* __restrict__ out,
    int N, int KNBR, int D) {
  const int C      = C1 + C2;
  const int nCt    = C >> 5;
  const int nPair  = D >> 4;       // # of 16-col a-tiles (== # g-tiles); always even here
  const int nColT  = nPair << 1;
  const int nPGrp  = nPair >> 1;   // column-pair groups of 2
  const int lane   = threadIdx.x & 31;
  const int wid    = threadIdx.x >> 5;
  const long tile  = (long)blockIdx.x * (blockDim.x >> 5) + wid;
  const int nRowT  = (N + 15) >> 4;
  if (tile >= (long)nRowT * nPGrp) return;     // wave-uniform exit
  const int rowT  = (int)(tile / nPGrp);
  const int pair0 = (int)(tile % nPGrp) * 2;   // handles col tiles pair0, pair0+1 (+gates)
  const int l16   = lane & 15;
  const int half  = lane >> 4;
  const int arow  = rowT * 16 + l16;           // row this lane gathers for A

  int par = -1, kp = -1;
  if (MODE == 1 && arow < N) { par = parents[arow]; kp = kposes[arow]; }

  f32x8 accA0 = {0.f,0.f,0.f,0.f,0.f,0.f,0.f,0.f};
  f32x8 accA1 = accA0, accG0 = accA0, accG1 = accA0;
  const u32x4 zq = {0u,0u,0u,0u};

  for (int k = 0; k < KNBR; ++k) {
    int r;
    if (MODE == 0) {
      r = (arow < N) ? nbr[(long)arow * KNBR + k] : -1;
    } else {
      if (__builtin_amdgcn_ballot_w32(kp == k) == 0u) continue;  // uniform skip
      r = (kp == k) ? par : -1;
    }
    for (int ct = 0; ct < nCt; ++ct) {
      int cb = ct << 5;
      const unsigned short* src = f1; int cs = C1;
      if (cb >= C1) { src = f2; cs = C2; cb -= C1; }
      ABits ab; ab.q[0] = zq; ab.q[1] = zq;
      if (r >= 0) {
        const unsigned short* p = src + (long)r * cs + cb + half * 8;
        ab.q[0] = *(const u32x4*)p;          // K-local {half*8 .. +7}
        ab.q[1] = *(const u32x4*)(p + 16);   // K-local {16+half*8 .. +7}
      }
      const long chunk = ((long)k * nCt + ct) * nColT;
      const unsigned short* bb = Bp + (chunk * 32 + lane) * 16;
      const bf16x16 bA0 = *(const bf16x16*)(bb + (long)(pair0        ) * 512);
      const bf16x16 bA1 = *(const bf16x16*)(bb + (long)(pair0 + 1    ) * 512);
      const bf16x16 bG0 = *(const bf16x16*)(bb + (long)(pair0 + nPair) * 512);
      const bf16x16 bG1 = *(const bf16x16*)(bb + (long)(pair0 + 1 + nPair) * 512);
      accA0 = __builtin_amdgcn_wmma_f32_16x16x32_bf16(false, ab.v, false, bA0, (short)0, accA0, false, false);
      accA1 = __builtin_amdgcn_wmma_f32_16x16x32_bf16(false, ab.v, false, bA1, (short)0, accA1, false, false);
      accG0 = __builtin_amdgcn_wmma_f32_16x16x32_bf16(false, ab.v, false, bG0, (short)0, accG0, false, false);
      accG1 = __builtin_amdgcn_wmma_f32_16x16x32_bf16(false, ab.v, false, bG1, (short)0, accG1, false, false);
    }
  }

  // Epilogue: gate + optional residual, store bf16.
  const int col0 = (pair0    ) * 16 + l16;
  const int col1 = (pair0 + 1) * 16 + l16;
  const int mb   = rowT * 16 + half * 8;
  if (rowT * 16 + 16 <= N) {
    // full tile: straight-line stores, no per-row predication
#pragma unroll
    for (int r2 = 0; r2 < 8; ++r2) {
      long m = mb + r2;
      float y0 = gate_f(accA0[r2], accG0[r2]);
      float y1 = gate_f(accA1[r2], accG1[r2]);
      if (resid) {
        y0 += bf2f(resid[m * D + col0]);
        y1 += bf2f(resid[m * D + col1]);
      }
      out[m * D + col0] = f2bf(y0);
      out[m * D + col1] = f2bf(y1);
    }
  } else {
#pragma unroll
    for (int r2 = 0; r2 < 8; ++r2) {
      int m = mb + r2;
      if (m < N) {
        float y0 = gate_f(accA0[r2], accG0[r2]);
        float y1 = gate_f(accA1[r2], accG1[r2]);
        if (resid) {
          y0 += bf2f(resid[(long)m * D + col0]);
          y1 += bf2f(resid[(long)m * D + col1]);
        }
        out[(long)m * D + col0] = f2bf(y0);
        out[(long)m * D + col1] = f2bf(y1);
      }
    }
  }
}

// Final linear: f32 out[n,32] = x[n,:32] @ Wlin + b  (single K=32 chunk,
// one wave covers both 16-col tiles)
__global__ __launch_bounds__(256) void linear_wmma(
    const unsigned short* __restrict__ x,
    const unsigned short* __restrict__ Bp,
    const float* __restrict__ bias,
    float* __restrict__ out, int N) {
  const int lane = threadIdx.x & 31;
  const int wid  = threadIdx.x >> 5;
  const long rowT = (long)blockIdx.x * (blockDim.x >> 5) + wid;
  const int nRowT = (N + 15) >> 4;
  if (rowT >= nRowT) return;
  const int l16  = lane & 15;
  const int half = lane >> 4;
  const int arow = (int)rowT * 16 + l16;

  ABits ab; const u32x4 zq = {0u,0u,0u,0u};
  ab.q[0] = zq; ab.q[1] = zq;
  if (arow < N) {
    const unsigned short* p = x + (long)arow * 32 + half * 8;
    ab.q[0] = *(const u32x4*)p;
    ab.q[1] = *(const u32x4*)(p + 16);
  }
  const bf16x16 b0 = *(const bf16x16*)(Bp + (long)lane * 16);
  const bf16x16 b1 = *(const bf16x16*)(Bp + (long)(32 + lane) * 16);
  f32x8 acc0 = {0.f,0.f,0.f,0.f,0.f,0.f,0.f,0.f};
  f32x8 acc1 = acc0;
  acc0 = __builtin_amdgcn_wmma_f32_16x16x32_bf16(false, ab.v, false, b0, (short)0, acc0, false, false);
  acc1 = __builtin_amdgcn_wmma_f32_16x16x32_bf16(false, ab.v, false, b1, (short)0, acc1, false, false);
  const int col0 = l16, col1 = 16 + l16;
  const int mb = (int)rowT * 16 + half * 8;
  const float bv0 = bias[col0], bv1 = bias[col1];
  if ((int)rowT * 16 + 16 <= N) {
#pragma unroll
    for (int r2 = 0; r2 < 8; ++r2) {
      long m = mb + r2;
      out[m * 32 + col0] = acc0[r2] + bv0;
      out[m * 32 + col1] = acc1[r2] + bv1;
    }
  } else {
#pragma unroll
    for (int r2 = 0; r2 < 8; ++r2) {
      int m = mb + r2;
      if (m < N) {
        out[(long)m * 32 + col0] = acc0[r2] + bv0;
        out[(long)m * 32 + col1] = acc1[r2] + bv1;
      }
    }
  }
}

// ---------------------------------------------------------------------------
// Host orchestration
// ---------------------------------------------------------------------------
static unsigned short* bump(char*& p, long elems) {
  uintptr_t a = (uintptr_t)p;
  a = (a + 255) & ~(uintptr_t)255;
  unsigned short* r = (unsigned short*)a;
  p = (char*)(a + (size_t)elems * 2);
  return r;
}

extern "C" void kernel_launch(void* const* d_in, const int* in_sizes, int n_in,
                              void* d_out, int out_size, void* d_ws, size_t ws_size,
                              hipStream_t stream) {
  (void)n_in; (void)out_size; (void)ws_size;
  const int P[5] = {32, 64, 96, 128, 160};

  const float* feats = (const float*)d_in[0];
  const float* w_in  = (const float*)d_in[1];
  const float* wr1[5]; const float* wr2[5];
  for (int l = 0; l < 5; ++l) { wr1[l] = (const float*)d_in[2 + l]; wr2[l] = (const float*)d_in[7 + l]; }
  const float* wd[4]; const float* wu[4]; const float* wdec[4];
  for (int l = 0; l < 4; ++l) {
    wd[l] = (const float*)d_in[12 + l];
    wu[l] = (const float*)d_in[16 + l];
    wdec[l] = (const float*)d_in[20 + l];
  }
  const float* wlin = (const float*)d_in[24];
  const float* blin = (const float*)d_in[25];
  const int* nbrs[5];   for (int l = 0; l < 5; ++l) nbrs[l]   = (const int*)d_in[26 + l];
  const int* childs[4]; for (int l = 0; l < 4; ++l) childs[l] = (const int*)d_in[31 + l];
  const int* parents[4];for (int l = 0; l < 4; ++l) parents[l]= (const int*)d_in[35 + l];
  const int* kposes[4]; for (int l = 0; l < 4; ++l) kposes[l] = (const int*)d_in[39 + l];

  int Nl[5];
  for (int l = 0; l < 5; ++l) Nl[l] = in_sizes[26 + l] / 27;

  char* wp = (char*)d_ws;

  auto repack = [&](const float* src, unsigned short* dst, int K, int Cs, int Cp, int TwoD) {
    long tot = (long)K * Cp * TwoD;
    int blocks = (int)((tot + 255) / 256);
    repack_w<<<blocks, 256, 0, stream>>>(src, dst, K, Cs, Cp, TwoD);
  };

  // --- pack weights to bf16 WMMA fragment layout ---
  unsigned short* Pin = bump(wp, 27L * 32 * 64);
  repack(w_in, Pin, 27, 16, 32, 64);
  unsigned short* Pr1[5]; unsigned short* Pr2[5];
  for (int l = 0; l < 5; ++l) {
    Pr1[l] = bump(wp, 27L * P[l] * 2 * P[l]); repack(wr1[l], Pr1[l], 27, P[l], P[l], 2 * P[l]);
    Pr2[l] = bump(wp, 27L * P[l] * 2 * P[l]); repack(wr2[l], Pr2[l], 27, P[l], P[l], 2 * P[l]);
  }
  unsigned short* Pd[4]; unsigned short* Pu[4]; unsigned short* Pdec[4];
  for (int l = 0; l < 4; ++l) {
    Pd[l]   = bump(wp, 8L  * P[l]     * 2 * P[l + 1]); repack(wd[l],   Pd[l],   8,  P[l],     P[l],     2 * P[l + 1]);
    Pu[l]   = bump(wp, 8L  * P[l + 1] * 2 * P[l]);     repack(wu[l],   Pu[l],   8,  P[l + 1], P[l + 1], 2 * P[l]);
    Pdec[l] = bump(wp, 27L * 2 * P[l] * 2 * P[l]);     repack(wdec[l], Pdec[l], 27, 2 * P[l], 2 * P[l], 2 * P[l]);
  }
  unsigned short* Plin = bump(wp, 32L * 32);
  repack(wlin, Plin, 1, 32, 32, 32);

  // --- activation buffers (bf16) ---
  unsigned short* xin = bump(wp, (long)Nl[0] * 32);
  {
    long tot = (long)Nl[0] * 32;
    cvt_pad<<<(int)((tot + 255) / 256), 256, 0, stream>>>(feats, xin, Nl[0], 16, 32);
  }
  unsigned short* xcur[5]; unsigned short* skip[5];
  xcur[0] = bump(wp, (long)Nl[0] * 32);
  for (int l = 1; l < 5; ++l) xcur[l] = bump(wp, (long)Nl[l] * P[l]);
  for (int l = 0; l < 5; ++l) skip[l] = bump(wp, (long)Nl[l] * P[l]);
  long maxNP = 0;
  for (int l = 0; l < 5; ++l) { long v = (long)Nl[l] * P[l]; if (v > maxNP) maxNP = v; }
  unsigned short* tbuf  = bump(wp, maxNP);
  unsigned short* upbuf = bump(wp, maxNP);
  unsigned short* dec[4];
  for (int l = 0; l < 4; ++l) dec[l] = bump(wp, (long)Nl[l] * P[l]);

  auto conv = [&](const unsigned short* f1, int C1, const unsigned short* f2, int C2,
                  const int* nb, const int* par, const int* kps,
                  const unsigned short* Bp, const unsigned short* resid,
                  unsigned short* out, int N, int K, int D, bool up) {
    int nRowT = (N + 15) / 16;
    long tiles = (long)nRowT * (D / 32);   // column-pair groups of 2
    int blocks = (int)((tiles + 7) / 8);   // 8 waves / block (256 thr)
    if (up)
      gconv_wmma<1><<<blocks, 256, 0, stream>>>(f1, C1, f2, C2, nb, par, kps, Bp, resid, out, N, K, D);
    else
      gconv_wmma<0><<<blocks, 256, 0, stream>>>(f1, C1, f2, C2, nb, par, kps, Bp, resid, out, N, K, D);
  };

  // --- encoder ---
  conv(xin, 32, nullptr, 0, nbrs[0], nullptr, nullptr, Pin, nullptr, xcur[0], Nl[0], 27, 32, false);
  for (int l = 0; l < 5; ++l) {
    conv(xcur[l], P[l], nullptr, 0, nbrs[l], nullptr, nullptr, Pr1[l], nullptr, tbuf,    Nl[l], 27, P[l], false);
    conv(tbuf,    P[l], nullptr, 0, nbrs[l], nullptr, nullptr, Pr2[l], xcur[l], skip[l], Nl[l], 27, P[l], false);
    if (l < 4)
      conv(skip[l], P[l], nullptr, 0, childs[l], nullptr, nullptr, Pd[l], nullptr, xcur[l + 1], Nl[l + 1], 8, P[l + 1], false);
  }

  // --- decoder ---
  const unsigned short* cur = skip[4];
  for (int l = 3; l >= 0; --l) {
    conv(cur, P[l + 1], nullptr, 0, nullptr, parents[l], kposes[l], Pu[l], nullptr, upbuf, Nl[l], 8, P[l], true);
    conv(skip[l], P[l], upbuf, P[l], nbrs[l], nullptr, nullptr, Pdec[l], nullptr, dec[l], Nl[l], 27, P[l], false);
    cur = dec[l];
  }

  // --- final linear (f32 output) ---
  {
    int nRowT = (Nl[0] + 15) / 16;
    int blocks = (nRowT + 7) / 8;
    linear_wmma<<<blocks, 256, 0, stream>>>(cur, Plin, blin, (float*)d_out, Nl[0]);
  }
}
